// NonLocalSparseAttention_70282844831833
// MI455X (gfx1250) — matrix operand order repaired and checked by reading, not compiled, not run on hardware
//
#include <hip/hip_runtime.h>
#include <hip/hip_bf16.h>

typedef __attribute__((ext_vector_type(16))) _Float16 v16h;
typedef __attribute__((ext_vector_type(8)))  float    v8f;
typedef unsigned int u32;

union AV { v16h v; u32 u[8]; uint4 q[2]; };

static __device__ inline v8f wmma_f16(v16h a, v16h b, v8f c) {
  return __builtin_amdgcn_wmma_f32_16x16x32_f16(false, a, false, b, (short)0, c, false, false);
}

// ---- CDNA5 async global<->LDS copies (ASYNCcnt-tracked, no VGPR roundtrip)
typedef __attribute__((address_space(3))) const void* lds_cvptr;
static __device__ inline u32 lds_addr(const void* p) {
  return (u32)(uintptr_t)(lds_cvptr)p;
}
static __device__ inline void async_ld_b128(u32 lds, const void* g) {
  asm volatile("global_load_async_to_lds_b128 %0, %1, off" :: "v"(lds), "v"(g) : "memory");
}
static __device__ inline void async_st_b128(void* g, u32 lds) {
  asm volatile("global_store_async_from_lds_b128 %0, %1, off" :: "v"(g), "v"(lds) : "memory");
}
static __device__ inline void wait_async0() {
  asm volatile("s_wait_asynccnt 0" ::: "memory");
}

#define Hh 96
#define Ww 96
#define CHN 256
#define CM 64
#define LL (Hh*Ww)          // 9216
#define NBATCH 4
#define NHASH 4
#define CHUNKSZ 144
#define NCHK 64             // LL / CHUNKSZ
#define TOT (NHASH*LL)      // 36864 per batch
#define NBLK 144            // TOT / 256

// ---------------------------------------------------------------- f32 -> f16
__global__ __launch_bounds__(256) void cvt_f16_kernel(const float* __restrict__ src,
                                                      _Float16* __restrict__ dst, int n) {
  int i = blockIdx.x * 256 + threadIdx.x;
  if (i < n) dst[i] = (_Float16)src[i];
}

// weights: (3,3,256,CO) f32  ->  [tap][co][kk] f16  (so LDS staging = memcpy)
__global__ __launch_bounds__(256) void cvt_w_kernel(const float* __restrict__ src,
                                                    _Float16* __restrict__ dst, int CO) {
  int i = blockIdx.x * 256 + threadIdx.x;     // over 9*CO*256
  int kk = i & 255; int rest = i >> 8; int co = rest % CO; int tap = rest / CO;
  if (tap < 9) dst[i] = (_Float16)src[((size_t)(tap * 256 + kk)) * CO + co];
}

// ------------------------------------------------------- implicit-GEMM conv
// block = 128 threads (4 waves). Tile: 16 pixels (one image row) x 64 out ch.
__global__ __launch_bounds__(128) void conv_wmma_kernel(
    const _Float16* __restrict__ inF, const _Float16* __restrict__ wFT,
    const float* __restrict__ bias, float* __restrict__ outF32,
    _Float16* __restrict__ outF16, int CO)
{
  __shared__ _Float16 inRows[3 * 18 * 256];   // 27648 B
  __shared__ _Float16 wT[64 * 256];           // 32768 B  [co_local][kk]

  const int tid  = threadIdx.x;
  const int lane = tid & 31;
  const int wv   = tid >> 5;           // wave 0..3
  const int xt   = blockIdx.x % (Ww / 16);
  const int coB  = (blockIdx.x / (Ww / 16)) * 64;
  const int y    = blockIdx.y;
  const int b    = blockIdx.z;
  const int x0   = xt * 16;

  // stage 3 input rows x 18 pixels x 256 ch (f16); zeros at borders
  for (int u = tid; u < 3 * 18 * 32; u += 128) {
    int c8 = u & 31; int px = (u >> 5) % 18; int row = u / (32 * 18);
    int yy = y + row - 1; int xx = x0 + px - 1;
    u32 la = lds_addr(inRows + (row * 18 + px) * 256 + c8 * 8);
    if (yy >= 0 && yy < Hh && xx >= 0 && xx < Ww)
      async_ld_b128(la, inF + ((size_t)(b * LL + yy * Ww + xx)) * CHN + c8 * 8);
    else
      *(uint4*)(inRows + (row * 18 + px) * 256 + c8 * 8) = make_uint4(0, 0, 0, 0);
  }
  wait_async0();

  v8f acc = {};
  const int ml = lane & 15;
  const int hi = lane >> 4;

  for (int tap = 0; tap < 9; ++tap) {
    const int dy = tap / 3, dx = tap % 3;
    __syncthreads();
    // stage weights for this tap: straight async copy (already transposed)
    for (int u = tid; u < 64 * 32; u += 128) {
      int row = u >> 5; int c16 = u & 31;
      async_ld_b128(lds_addr(wT + row * 256 + c16 * 8),
                    wFT + ((size_t)(tap * CO + coB + row)) * 256 + c16 * 8);
    }
    wait_async0();
    __syncthreads();
    const uint4* ar = (const uint4*)(inRows + (dy * 18 + (ml + dx)) * 256);
    const uint4* br = (const uint4*)(wT + (wv * 16 + ml) * 256);
    for (int ks = 0; ks < 8; ++ks) {
      AV a, bb;
      a.q[0]  = ar[ks * 4 + hi];
      a.q[1]  = ar[ks * 4 + 2 + hi];
      bb.q[0] = br[ks * 4 + hi];
      bb.q[1] = br[ks * 4 + 2 + hi];
      acc = wmma_f16(a.v, bb.v, acc);
    }
  }

  const int co = coB + wv * 16 + ml;
  float bv = bias[co];
  #pragma unroll
  for (int r = 0; r < 8; ++r) {
    int mrow = r + hi * 8;
    int t = y * Ww + x0 + mrow;
    float val = acc[r] + bv;
    size_t o = ((size_t)(b * LL + t)) * CO + co;
    if (outF32) outF32[o] = val;
    outF16[o] = (_Float16)val;
  }
}

// ----------------------------------------------------------------- hashing
__global__ __launch_bounds__(256) void hash_kernel(const float* __restrict__ xm,
                                                   const float* __restrict__ rot,
                                                   int* __restrict__ codes)
{
  __shared__ float R[64 * 4 * 32];
  for (int u = threadIdx.x; u < 8192; u += 256) R[u] = rot[u];
  __syncthreads();
  int g = blockIdx.x * 256 + threadIdx.x;   // 0..N*L-1
  int b = g / LL; int t = g % LL;
  float x[64];
  const float* xp = xm + (size_t)(b * LL + t) * CM;
  #pragma unroll
  for (int f = 0; f < 64; ++f) x[f] = xp[f];
  for (int h = 0; h < NHASH; ++h) {
    float best = -1e30f; int bi = 0;
    for (int i = 0; i < 32; ++i) {
      float s = 0.f;
      #pragma unroll
      for (int f = 0; f < 64; ++f) s += x[f] * R[f * 128 + h * 32 + i];
      if (s > best)  { best = s;  bi = i; }
      if (-s > best) { best = -s; bi = 32 + i; }
    }
    codes[(size_t)b * TOT + h * LL + t] = bi + h * 64;
  }
}

// ----------------------------------------------- deterministic counting sort
__global__ __launch_bounds__(256) void hist_kernel(const int* __restrict__ codes,
                                                   int* __restrict__ blockHist)
{
  __shared__ int hcnt[256];
  int b = blockIdx.y, blk = blockIdx.x;
  hcnt[threadIdx.x] = 0;
  __syncthreads();
  int c = codes[(size_t)b * TOT + blk * 256 + threadIdx.x];
  atomicAdd(&hcnt[c], 1);
  __syncthreads();
  blockHist[((size_t)b * 256 + threadIdx.x) * NBLK + blk] = hcnt[threadIdx.x];
}

__global__ __launch_bounds__(256) void scan_kernel(const int* __restrict__ blockHist,
                                                   int* __restrict__ blockOff)
{
  __shared__ int tot[256];
  __shared__ int start[256];
  int b = blockIdx.x; int u = threadIdx.x;
  const int* bh = blockHist + (size_t)b * 256 * NBLK;
  int s = 0;
  for (int blk = 0; blk < NBLK; ++blk) s += bh[u * NBLK + blk];
  tot[u] = s;
  __syncthreads();
  if (u == 0) { int run = 0; for (int i = 0; i < 256; ++i) { start[i] = run; run += tot[i]; } }
  __syncthreads();
  int run = start[u];
  int* bo = blockOff + (size_t)b * 256 * NBLK;
  for (int blk = 0; blk < NBLK; ++blk) { bo[u * NBLK + blk] = run; run += bh[u * NBLK + blk]; }
}

__global__ __launch_bounds__(256) void scatter_kernel(const int* __restrict__ codes,
                                                      const int* __restrict__ blockOff,
                                                      int* __restrict__ idxA,
                                                      int* __restrict__ undoA)
{
  __shared__ int sc[256];
  int b = blockIdx.y, blk = blockIdx.x, i = threadIdx.x;
  int s = blk * 256 + i;
  int c = codes[(size_t)b * TOT + s];
  sc[i] = c;
  __syncthreads();
  int rank = 0;
  for (int j = 0; j < i; ++j) rank += (sc[j] == c);   // stable within block
  int pos = blockOff[((size_t)b * 256 + c) * NBLK + blk] + rank;
  idxA[(size_t)b * TOT + pos] = s;
  undoA[(size_t)b * TOT + s] = pos;
}

// --------------------------------------------------- chunked WMMA attention
// block = 288 threads (9 waves), one block per (b,h,chunk). Dynamic LDS:
//   XK     [448][64] f16   57344 B   (gathered x rows: chunk|prev|next; zero-pad)
//   scaleA [448] f32        1792 B
//   rowLse [144] f32         576 B
//   idxmod [432] int        1728 B
//   VS     [32][256] f16   16384 B   (V k-slab, row-major, async staged)
//   Pst    [9][16][32] f16  9216 B   (per-wave P slab for A-layout reload)
// epilogue reuses bytes [0, 73728) as the 144x256 f16 output staging buffer.
#define ATTN_LDS_BYTES 87040
__global__ __launch_bounds__(288) void attn_kernel(
    const _Float16* __restrict__ xmF, const _Float16* __restrict__ yaF,
    const int* __restrict__ idxA, _Float16* __restrict__ retS,
    float* __restrict__ bsS)
{
  extern __shared__ char smemRaw[];
  _Float16* XK    = (_Float16*)smemRaw;                 // 448*64
  float*    scaleA= (float*)(smemRaw + 57344);
  float*    rowLse= (float*)(smemRaw + 59136);
  int*      idxmod= (int*)(smemRaw + 59712);
  _Float16* VS    = (_Float16*)(smemRaw + 61440);       // [32][256]
  _Float16* Pst   = (_Float16*)(smemRaw + 77824);       // [9][16][32]

  const int tid  = threadIdx.x;
  const int lane = tid & 31;
  const int wv   = tid >> 5;     // 0..8
  const int k    = blockIdx.x;
  const int h    = blockIdx.y;
  const int b    = blockIdx.z;
  const size_t ibase = (size_t)b * TOT;

  // 1) source indices for rows of [this chunk | prev chunk | next chunk]
  if (tid < 432) {
    int seg = tid / CHUNKSZ;
    int jj  = tid % CHUNKSZ;
    int kk  = (seg == 0) ? k : (seg == 1 ? (k + NCHK - 1) % NCHK : (k + 1) % NCHK);
    idxmod[tid] = idxA[ibase + h * LL + kk * CHUNKSZ + jj] % LL;
  }
  __syncthreads();

  // 2) gather XK rows via async copies; zero rows 432..447
  for (int u = tid; u < 448 * 8; u += 288) {
    int row = u >> 3; int c8 = u & 7;
    if (row < 432) {
      int t = idxmod[row];
      async_ld_b128(lds_addr(XK + row * 64 + c8 * 8),
                    xmF + ((size_t)(b * LL + t)) * CM + c8 * 8);
    } else {
      *(uint4*)(XK + row * 64 + c8 * 8) = make_uint4(0, 0, 0, 0);
    }
  }
  wait_async0();
  __syncthreads();

  // 3) key-side L2 normalization folded as a per-column scale
  for (int row = tid; row < 448; row += 288) {
    float ss = 0.f;
    const _Float16* xr = XK + row * 64;
    for (int e = 0; e < 64; ++e) { float v = (float)xr[e]; ss += v * v; }
    scaleA[row] = (row < 432) ? rsqrtf(fmaxf(ss, 5e-5f)) : 0.f;
  }
  __syncthreads();

  const int ml = lane & 15;
  const int hi = lane >> 4;
  const int r0 = wv * 16;

  // A-operand (Q rows r0..r0+15), two K=32 steps covering 64 features
  AV aQ0, aQ1;
  {
    const uint4* ar = (const uint4*)(XK + (r0 + ml) * 64);
    aQ0.q[0] = ar[hi * 2];     aQ0.q[1] = ar[hi * 2 + 1];
    aQ1.q[0] = ar[4 + hi * 2]; aQ1.q[1] = ar[5 + hi * 2];
  }

  // pass 1: online per-lane (max, sumexp) over all 27 key tiles
  float m8[8], s8[8];
  #pragma unroll
  for (int r = 0; r < 8; ++r) { m8[r] = -1e30f; s8[r] = 0.f; }
  for (int q = 0; q < 27; ++q) {
    int j0 = q * 16;
    const uint4* br = (const uint4*)(XK + (j0 + ml) * 64);
    AV b0, b1;
    b0.q[0] = br[hi * 2];     b0.q[1] = br[hi * 2 + 1];
    b1.q[0] = br[4 + hi * 2]; b1.q[1] = br[5 + hi * 2];
    v8f racc = {};
    racc = wmma_f16(aQ0.v, b0.v, racc);
    racc = wmma_f16(aQ1.v, b1.v, racc);
    float sc = scaleA[j0 + ml];
    #pragma unroll
    for (int r = 0; r < 8; ++r) {
      float x = racc[r] * sc;
      float nm = fmaxf(m8[r], x);
      s8[r] = s8[r] * __expf(m8[r] - nm) + __expf(x - nm);
      m8[r] = nm;
    }
  }
  // 16-lane shuffle reduction -> lse per row
  float lse8[8];
  #pragma unroll
  for (int r = 0; r < 8; ++r) {
    float m = m8[r], s = s8[r];
    #pragma unroll
    for (int off = 1; off < 16; off <<= 1) {
      float om = __shfl_xor(m, off, 32);
      float os = __shfl_xor(s, off, 32);
      float nm = fmaxf(m, om);
      s = s * __expf(m - nm) + os * __expf(om - nm);
      m = nm;
    }
    lse8[r] = m + __logf(s);
  }
  if (ml == 0) {
    #pragma unroll
    for (int r = 0; r < 8; ++r) rowLse[r0 + hi * 8 + r] = lse8[r];
  }

  // pass 2: recompute P slab-by-slab (K=32), stream V through LDS, PV WMMA
  v8f o[16];
  {
    v8f z = {};
    #pragma unroll
    for (int ct = 0; ct < 16; ++ct) o[ct] = z;
  }
  for (int ks = 0; ks < 14; ++ks) {
    int j0 = ks * 32;
    __syncthreads();   // previous slab fully consumed
    // P tiles for this slab (C-layout -> Pst)
    #pragma unroll
    for (int qq = 0; qq < 2; ++qq) {
      int j1 = j0 + qq * 16;
      if (j1 < 432) {
        const uint4* br = (const uint4*)(XK + (j1 + ml) * 64);
        AV b0, b1;
        b0.q[0] = br[hi * 2];     b0.q[1] = br[hi * 2 + 1];
        b1.q[0] = br[4 + hi * 2]; b1.q[1] = br[5 + hi * 2];
        v8f racc = {};
        racc = wmma_f16(aQ0.v, b0.v, racc);
        racc = wmma_f16(aQ1.v, b1.v, racc);
        float sc = scaleA[j1 + ml];
        #pragma unroll
        for (int r = 0; r < 8; ++r) {
          float p = __expf(racc[r] * sc - lse8[r]);
          Pst[(wv * 16 + r + hi * 8) * 32 + qq * 16 + ml] = (_Float16)p;
        }
      } else {
        #pragma unroll
        for (int r = 0; r < 8; ++r)
          Pst[(wv * 16 + r + hi * 8) * 32 + qq * 16 + ml] = (_Float16)0.f;
      }
    }
    // stage V slab (row-major) via async copies
    for (int u = tid; u < 32 * 32; u += 288) {
      int jr = u >> 5; int c16 = u & 31;
      int row = j0 + jr;
      if (row < 432) {
        int t = idxmod[row];
        async_ld_b128(lds_addr(VS + jr * 256 + c16 * 8),
                      yaF + ((size_t)(b * LL + t)) * CHN + c16 * 8);
      } else {
        *(uint4*)(VS + jr * 256 + c16 * 8) = make_uint4(0, 0, 0, 0);
      }
    }
    wait_async0();
    __syncthreads();
    // PV: A = P slab (A-layout reload), B = V columns (u16 pair loads)
    AV aP;
    {
      const uint4* pr = (const uint4*)(Pst + (wv * 16 + ml) * 32);
      aP.q[0] = pr[hi * 2];
      aP.q[1] = pr[hi * 2 + 1];
    }
    #pragma unroll
    for (int ct = 0; ct < 16; ++ct) {
      int c = ct * 16 + ml;
      AV bV;
      #pragma unroll
      for (int v = 0; v < 8; ++v) {
        u32 lo = *(const unsigned short*)(VS + (hi * 16 + 2 * v) * 256 + c);
        u32 hi2 = *(const unsigned short*)(VS + (hi * 16 + 2 * v + 1) * 256 + c);
        bV.u[v] = lo | (hi2 << 16);
      }
      o[ct] = wmma_f16(aP.v, bV.v, o[ct]);
    }
  }
  __syncthreads();

  // lse outputs, then reuse LDS as coalescing buffer for ret
  if (tid < CHUNKSZ) bsS[ibase + h * LL + k * CHUNKSZ + tid] = rowLse[tid];
  __syncthreads();
  _Float16* S = (_Float16*)smemRaw;     // 144*256 f16 = 73728 B
  #pragma unroll
  for (int ct = 0; ct < 16; ++ct) {
    int c = ct * 16 + ml;
    #pragma unroll
    for (int r = 0; r < 8; ++r)
      S[(r0 + r + hi * 8) * 256 + c] = (_Float16)o[ct][r];
  }
  __syncthreads();
  {
    size_t obase = (ibase + (size_t)(h * LL + k * CHUNKSZ)) * CHN;
    for (int u = tid; u < CHUNKSZ * 32; u += 288) {
      int i = u >> 5; int c16 = u & 31;
      async_st_b128(retS + obase + (size_t)i * CHN + c16 * 8,
                    lds_addr(S + i * 256 + c16 * 8));
    }
    wait_async0();
  }
}

// ----------------------------------------- unsort + hash-softmax + residual
__global__ __launch_bounds__(256) void combine_kernel(
    const float* __restrict__ inp, const _Float16* __restrict__ retS,
    const float* __restrict__ bsS, const int* __restrict__ undoA,
    float* __restrict__ out)
{
  __shared__ float pr[4];
  __shared__ int   pos[4];
  int t = blockIdx.x; int b = blockIdx.y;
  size_t ibase = (size_t)b * TOT;
  if (threadIdx.x == 0) {
    float bsv[4]; int p4[4];
    float m = -1e30f;
    for (int hh = 0; hh < NHASH; ++hh) {
      p4[hh] = undoA[ibase + hh * LL + t];
      bsv[hh] = bsS[ibase + p4[hh]];
      m = fmaxf(m, bsv[hh]);
    }
    float ssum = 0.f;
    for (int hh = 0; hh < NHASH; ++hh) { bsv[hh] = __expf(bsv[hh] - m); ssum += bsv[hh]; }
    for (int hh = 0; hh < NHASH; ++hh) { pr[hh] = bsv[hh] / ssum; pos[hh] = p4[hh]; }
  }
  __syncthreads();
  int c = threadIdx.x;
  float accv = 0.f;
  for (int hh = 0; hh < NHASH; ++hh)
    accv += pr[hh] * (float)retS[(ibase + (size_t)pos[hh]) * CHN + c];
  size_t oidx = ((size_t)(b * LL + t)) * CHN + c;
  out[oidx] = accv * 0.1f + inp[oidx];
}

// --------------------------------------------------------------------------
extern "C" void kernel_launch(void* const* d_in, const int* in_sizes, int n_in,
                              void* d_out, int out_size, void* d_ws, size_t ws_size,
                              hipStream_t stream) {
  (void)in_sizes; (void)n_in; (void)out_size; (void)ws_size;
  const float* input     = (const float*)d_in[0];
  const float* w_match   = (const float*)d_in[1];
  const float* b_match   = (const float*)d_in[2];
  const float* w_asm     = (const float*)d_in[3];
  const float* b_asm     = (const float*)d_in[4];
  const float* rotations = (const float*)d_in[5];
  float* out = (float*)d_out;

  char* ws = (char*)d_ws;
  size_t off = 0;
  auto carve = [&](size_t bytes) -> void* {
    void* p = ws + off;
    off += (bytes + 255) & ~(size_t)255;
    return p;
  };
  const int nIn = NBATCH * LL * CHN;           // 9,437,184
  const int nWm = 9 * CHN * CM;                // 147,456
  const int nWa = 9 * CHN * CHN;               // 589,824

  _Float16* inF   = (_Float16*)carve((size_t)nIn * 2);
  _Float16* wmFT  = (_Float16*)carve((size_t)nWm * 2);
  _Float16* waFT  = (_Float16*)carve((size_t)nWa * 2);
  float*    xmF32 = (float*)   carve((size_t)NBATCH * LL * CM * 4);
  _Float16* xmF16 = (_Float16*)carve((size_t)NBATCH * LL * CM * 2);
  _Float16* yaF16 = (_Float16*)carve((size_t)NBATCH * LL * CHN * 2);
  int*      codes = (int*)     carve((size_t)NBATCH * TOT * 4);
  int*      idxA  = (int*)     carve((size_t)NBATCH * TOT * 4);
  int*      undoA = (int*)     carve((size_t)NBATCH * TOT * 4);
  int*      bHist = (int*)     carve((size_t)NBATCH * 256 * NBLK * 4);
  int*      bOff  = (int*)     carve((size_t)NBATCH * 256 * NBLK * 4);
  _Float16* retS  = (_Float16*)carve((size_t)NBATCH * TOT * CHN * 2);
  float*    bsS   = (float*)   carve((size_t)NBATCH * TOT * 4);

  // 0) f16 conversions (weights transposed to [tap][co][kk])
  cvt_f16_kernel<<<(nIn + 255) / 256, 256, 0, stream>>>(input, inF, nIn);
  cvt_w_kernel<<<(nWm + 255) / 256, 256, 0, stream>>>(w_match, wmFT, CM);
  cvt_w_kernel<<<(nWa + 255) / 256, 256, 0, stream>>>(w_asm, waFT, CHN);

  // 1) convs (implicit GEMM, WMMA)
  conv_wmma_kernel<<<dim3((Ww / 16) * (CM / 64), Hh, NBATCH), 128, 0, stream>>>(
      inF, wmFT, b_match, xmF32, xmF16, CM);
  conv_wmma_kernel<<<dim3((Ww / 16) * (CHN / 64), Hh, NBATCH), 128, 0, stream>>>(
      inF, waFT, b_asm, nullptr, yaF16, CHN);

  // 2) LSH codes
  hash_kernel<<<(NBATCH * LL) / 256, 256, 0, stream>>>(xmF32, rotations, codes);

  // 3) deterministic stable counting sort (codes in [0,256))
  hist_kernel<<<dim3(NBLK, NBATCH), 256, 0, stream>>>(codes, bHist);
  scan_kernel<<<NBATCH, 256, 0, stream>>>(bHist, bOff);
  scatter_kernel<<<dim3(NBLK, NBATCH), 256, 0, stream>>>(codes, bOff, idxA, undoA);

  // 4) chunked attention (one block per (b,h,chunk))
  attn_kernel<<<dim3(NCHK, NHASH, NBATCH), 288, ATTN_LDS_BYTES, stream>>>(
      xmF16, yaF16, idxA, retS, bsS);

  // 5) unsort + softmax over hashes + residual
  combine_kernel<<<dim3(LL, NBATCH), 256, 0, stream>>>(input, retS, bsS, undoA, out);
}
